// MessagePassing_40750649705199
// MI455X (gfx1250) — matrix operand order repaired and checked by reading, not compiled, not run on hardware
//
#include <hip/hip_runtime.h>

typedef __attribute__((ext_vector_type(16))) _Float16 v16h;
typedef __attribute__((ext_vector_type(8)))  _Float16 hv8;
typedef __attribute__((ext_vector_type(8)))  float    v8f;

#define D64 64

__device__ __forceinline__ v8f wmma_f16(v16h a, v16h b, v8f c) {
    // D = A(16x32 f16) x B(32x16 f16) + C(16x16 f32)
    return __builtin_amdgcn_wmma_f32_16x16x32_f16(
        /*neg_a=*/false, a, /*neg_b=*/false, b,
        /*c_mod=*/(short)0, c, /*reuse_a=*/false, /*reuse_b=*/false);
}

__device__ __forceinline__ v8f lrelu(v8f x) {
#pragma unroll
    for (int i = 0; i < 8; ++i) x[i] = x[i] >= 0.f ? x[i] : 0.01f * x[i];
    return x;
}

// Device-scope non-returning f32 atomic add: guaranteed single hardware
// global_atomic_add_f32 (RMW resolves at L2, the device coherence point).
__device__ __forceinline__ void atomic_add_f32_dev(float* p, float v) {
    asm volatile("global_atomic_add_f32 %0, %1, off scope:SCOPE_DEV"
                 :: "v"(p), "v"(v) : "memory");
}

// Load one 16x32 f16 A-chunk from an fp32 row, in contiguous k order (same order as the B loads):
// lane l (row M = l&15): halfs 0..15 -> K = c*32 + 2*kh + i
__device__ __forceinline__ v16h load_a_f32row(const float* __restrict__ rowp, int c, int kh) {
    const float4* p = reinterpret_cast<const float4*>(rowp + c * 32 + 2 * kh);
    const float4* q = reinterpret_cast<const float4*>(rowp + c * 32 + 2 * kh + 8);
    float4 x0 = p[0], x1 = p[1], y0 = q[0], y1 = q[1];
    v16h a;
    a[0]  = (_Float16)x0.x; a[1]  = (_Float16)x0.y; a[2]  = (_Float16)x0.z; a[3]  = (_Float16)x0.w;
    a[4]  = (_Float16)x1.x; a[5]  = (_Float16)x1.y; a[6]  = (_Float16)x1.z; a[7]  = (_Float16)x1.w;
    a[8]  = (_Float16)y0.x; a[9]  = (_Float16)y0.y; a[10] = (_Float16)y0.z; a[11] = (_Float16)y0.w;
    a[12] = (_Float16)y1.x; a[13] = (_Float16)y1.y; a[14] = (_Float16)y1.z; a[15] = (_Float16)y1.w;
    return a;
}

// ---------------- prep kernels ----------------

__global__ __launch_bounds__(256) void zero_out_kernel(float* __restrict__ out, int n) {
    int i = blockIdx.x * 256 + threadIdx.x;
    if (i < n) out[i] = 0.f;
}

__global__ __launch_bounds__(256) void prep_weights_kernel(
    const float* __restrict__ Wn, const float* __restrict__ W1, const float* __restrict__ W2,
    _Float16* __restrict__ WnT, _Float16* __restrict__ W1T, _Float16* __restrict__ W2T) {
    int i = blockIdx.x * 256 + threadIdx.x;  // i = n*64 + k  (transposed layout [n][k])
    if (i < 64 * 64) {
        int n = i >> 6, k = i & 63;
        int src = k * 64 + n;
        WnT[i] = (_Float16)Wn[src];
        W1T[i] = (_Float16)W1[src];
        W2T[i] = (_Float16)W2[src];
    }
}

// ---------------- node projection: m = node @ Wn ----------------

__global__ __launch_bounds__(256) void node_proj_kernel(
    const float* __restrict__ node, const _Float16* __restrict__ WnT,
    float* __restrict__ mtab, int numTiles) {
    __shared__ alignas(32) _Float16 wlds[64 * 64];
    for (int i = threadIdx.x; i < 64 * 64; i += 256) wlds[i] = WnT[i];
    __syncthreads();

    const int lane = threadIdx.x & 31;
    const int wid  = threadIdx.x >> 5;
    const int l15  = lane & 15;
    const int hi   = lane >> 4;
    const int kh   = hi * 8;

    int tile = blockIdx.x * 8 + wid;
    if (tile >= numTiles) return;
    int base = tile * 16;

    v16h a0 = load_a_f32row(node + (size_t)(base + l15) * D64, 0, kh);
    v16h a1 = load_a_f32row(node + (size_t)(base + l15) * D64, 1, kh);

#pragma unroll
    for (int t = 0; t < 4; ++t) {
        // B layout (32x16 f16): lane l -> column N = l&15; halfs 0..15 -> K = c*32 + hi*16 + i
        const v16h* bp = reinterpret_cast<const v16h*>(wlds) + (t * 16 + l15) * 4;
        v8f d = {};
        d = wmma_f16(a0, bp[hi],     d);   // K = 0..31
        d = wmma_f16(a1, bp[2 + hi], d);   // K = 32..63
        // C/D layout: lane l -> (M = r + 8*hi, N = l&15)
#pragma unroll
        for (int r = 0; r < 8; ++r)
            mtab[(size_t)(base + r + 8 * hi) * D64 + t * 16 + l15] = d[r];
    }
}

// ---------------- edge MLP + gather-multiply + scatter-add ----------------

__global__ __launch_bounds__(256) void edge_mp_kernel(
    const float* __restrict__ edge, const int* __restrict__ seg_i,
    const int* __restrict__ idx_j, const float* __restrict__ b1,
    const float* __restrict__ b2, const _Float16* __restrict__ W1T,
    const _Float16* __restrict__ W2T, const float* __restrict__ mtab,
    float* __restrict__ out, int numTiles) {
    __shared__ alignas(32) _Float16 wlds[2 * 64 * 64];   // W1T | W2T      (16 KB)
    __shared__ alignas(32) _Float16 hlds[8 * 16 * 64];   // per-wave h1    (16 KB)

    for (int i = threadIdx.x; i < 64 * 64; i += 256) {
        wlds[i]           = W1T[i];
        wlds[64 * 64 + i] = W2T[i];
    }
    __syncthreads();

    const int lane = threadIdx.x & 31;
    const int wid  = threadIdx.x >> 5;
    const int l15  = lane & 15;
    const int hi   = lane >> 4;
    const int kh   = hi * 8;

    int tile = blockIdx.x * 8 + wid;
    bool active = tile < numTiles;   // wave-uniform

    int myj = 0, mys = 0;
    if (active) {
        int base = tile * 16;
        const float* rowp = edge + (size_t)(base + l15) * D64;
        v16h a0 = load_a_f32row(rowp, 0, kh);
        v16h a1 = load_a_f32row(rowp, 1, kh);
        myj = idx_j[base + l15];
        mys = seg_i[base + l15];

        _Float16* hw = hlds + wid * 16 * 64;
#pragma unroll
        for (int t = 0; t < 4; ++t) {
            float bv = b1[t * 16 + l15];
            v8f d;
#pragma unroll
            for (int i = 0; i < 8; ++i) d[i] = bv;   // bias pre-loaded into C
            const v16h* bp = reinterpret_cast<const v16h*>(wlds) + (t * 16 + l15) * 4;
            d = wmma_f16(a0, bp[hi],     d);
            d = wmma_f16(a1, bp[2 + hi], d);
            d = lrelu(d);
            // h1 in C/D layout -> LDS row-major [M][64] f16
#pragma unroll
            for (int r = 0; r < 8; ++r)
                hw[(r + 8 * hi) * 64 + t * 16 + l15] = (_Float16)d[r];
        }
    }
    __syncthreads();   // make intra-wave cross-lane LDS traffic visible

    if (active) {
        int base = tile * 16;
        const _Float16* hw = hlds + wid * 16 * 64;
        // Re-load h1 as A-matrix chunks (A layout): two 8-half segments per chunk
        const hv8* hp = reinterpret_cast<const hv8*>(hw + l15 * 64);
        hv8 c0lo = hp[2 * hi],     c0hi = hp[2 * hi + 1];   // K = 16*hi..16*hi+15
        hv8 c1lo = hp[4 + 2 * hi], c1hi = hp[5 + 2 * hi];   // K = 32+16*hi..32+16*hi+15
        v16h h0 = __builtin_shufflevector(c0lo, c0hi, 0,1,2,3,4,5,6,7,8,9,10,11,12,13,14,15);
        v16h h1 = __builtin_shufflevector(c1lo, c1hi, 0,1,2,3,4,5,6,7,8,9,10,11,12,13,14,15);

#pragma unroll
        for (int t = 0; t < 4; ++t) {
            float bv = b2[t * 16 + l15];
            v8f d;
#pragma unroll
            for (int i = 0; i < 8; ++i) d[i] = bv;
            const v16h* bp = reinterpret_cast<const v16h*>(wlds + 64 * 64) + (t * 16 + l15) * 4;
            d = wmma_f16(h0, bp[hi],     d);
            d = wmma_f16(h1, bp[2 + hi], d);
            d = lrelu(d);

            int col = t * 16 + l15;
#pragma unroll
            for (int r = 0; r < 8; ++r) {
                int j = __shfl(myj, r + 8 * hi, 32);   // edge row M = r + 8*hi of this tile
                int s = __shfl(mys, r + 8 * hi, 32);
                float val = d[r] * mtab[(size_t)j * D64 + col];
                atomic_add_f32_dev(out + (size_t)s * D64 + col, val);
            }
        }
    }
}

// ---------------- launch ----------------

extern "C" void kernel_launch(void* const* d_in, const int* in_sizes, int n_in,
                              void* d_out, int out_size, void* d_ws, size_t ws_size,
                              hipStream_t stream) {
    const float* node  = (const float*)d_in[0];
    const float* edge  = (const float*)d_in[1];
    const int*   seg_i = (const int*)d_in[2];
    const int*   idx_j = (const int*)d_in[3];
    const float* Wn    = (const float*)d_in[4];
    const float* W1    = (const float*)d_in[5];
    const float* b1    = (const float*)d_in[6];
    const float* W2    = (const float*)d_in[7];
    const float* b2    = (const float*)d_in[8];
    float* out = (float*)d_out;

    const int N = in_sizes[0] / D64;   // 100000
    const int E = in_sizes[1] / D64;   // 1600000

    // workspace: [ mtab f32 N*64 | WnT f16 4096 | W1T f16 4096 | W2T f16 4096 ]
    float*    mtab = (float*)d_ws;
    _Float16* WnT  = (_Float16*)((char*)d_ws + (size_t)N * D64 * sizeof(float));
    _Float16* W1T  = WnT + 64 * 64;
    _Float16* W2T  = W1T + 64 * 64;

    int nOut = N * D64;
    zero_out_kernel<<<(nOut + 255) / 256, 256, 0, stream>>>(out, nOut);
    prep_weights_kernel<<<(64 * 64 + 255) / 256, 256, 0, stream>>>(Wn, W1, W2, WnT, W1T, W2T);

    int nTilesN = N / 16;              // 6250 (N multiple of 16)
    node_proj_kernel<<<(nTilesN + 7) / 8, 256, 0, stream>>>(node, WnT, mtab, nTilesN);

    int nTilesE = E / 16;              // 100000 (E multiple of 16)
    edge_mp_kernel<<<(nTilesE + 7) / 8, 256, 0, stream>>>(
        edge, seg_i, idx_j, b1, b2, W1T, W2T, mtab, out, nTilesE);
}
